// PeriodicConvOp_43258910605688
// MI455X (gfx1250) — matrix operand order repaired and checked by reading, not compiled, not run on hardware
//
#include <hip/hip_runtime.h>
#include <hip/hip_bf16.h>

// PeriodicConv (period 2, 3x3) as 4 polyphase GEMMs, tap-major K ordering.
// Pipeline:
//   prep kernel : x (f32, NCHW) -> xT (bf16, [b][h][w][c] channel-minor) in d_ws
//                 W (f32)       -> Wt2 (bf16, tap-major, K padded 144->160) in d_ws
//   main kernel : per block, TDM tensor_load_to_lds pulls the 17x(33x16) bf16
//                 halo tile into LDS (border blocks: manual b128 copies), then
//                 v_wmma_f32_16x16x32_bf16 with B fragments read directly from
//                 the raw tile (no im2col materialization).
// Fallback kernel (if ws too small): round-2 self-contained version.

typedef __attribute__((ext_vector_type(16))) __bf16 v16bf;
typedef __attribute__((ext_vector_type(8)))  float  v8f;
typedef unsigned short u16;
typedef __attribute__((ext_vector_type(8)))  u16 v8u;
typedef __attribute__((ext_vector_type(16))) u16 v16u;
typedef __attribute__((ext_vector_type(4)))  unsigned int u32x4;
typedef __attribute__((ext_vector_type(4)))  int i32x4;
typedef __attribute__((ext_vector_type(8)))  int i32x8;

#define BATCH 8
#define CIN   16
#define HH    256
#define WW    256
#define COUT  64
#define KC    144          // 9 taps * 16 channels
#define KPAD  160          // 10 taps (tap 9 = zero pad)
#define HT    8            // hp tile
#define WT    16           // wp tile
#define RAWH  17           // 2*HT + 1 halo rows
#define RAWW  33           // 2*WT + 1 halo cols

__device__ __forceinline__ u16 f32_to_bf16_rne(float f) {
  unsigned u = __float_as_uint(f);
  unsigned r = u + 0x7FFFu + ((u >> 16) & 1u);
  return (u16)(r >> 16);
}

__device__ __forceinline__ v16bf frag_cast(v8u lo, v8u hi) {
  union { v16u u; v16bf b; } cvt;
  cvt.u = __builtin_shufflevector(lo, hi, 0, 1, 2, 3, 4, 5, 6, 7,
                                          8, 9, 10, 11, 12, 13, 14, 15);
  return cvt.b;
}

// ---------------------------------------------------------------------------
// Prep: transpose/convert x to channel-minor bf16; reorder weights tap-major.
// ---------------------------------------------------------------------------
__global__ __launch_bounds__(256) void periodic_conv_prep_kernel(
    const float* __restrict__ x, const float* __restrict__ Wm,
    u16* __restrict__ xT, u16* __restrict__ Wt2)
{
  if (blockIdx.x == BATCH * HH) {   // last block: weights
    for (int i = threadIdx.x; i < 4 * COUT * KPAD; i += 256) {
      const int gco = i / KPAD;
      const int k   = i - gco * KPAD;
      const int t   = k >> 4;
      const int c   = k & 15;
      float v = (t < 9) ? Wm[gco * KC + c * 9 + t] : 0.0f;
      Wt2[i] = f32_to_bf16_rne(v);
    }
    return;
  }
  const int bh = blockIdx.x;              // b*HH + h
  const int w  = threadIdx.x;             // one pixel column per thread
  const float* __restrict__ src = x + ((size_t)(bh >> 8) * CIN * HH + (bh & 255)) * WW + w;
  unsigned packed[8];
#pragma unroll
  for (int cp = 0; cp < 8; ++cp) {
    const u16 lo = f32_to_bf16_rne(src[(size_t)(2 * cp) * HH * WW]);
    const u16 hi = f32_to_bf16_rne(src[(size_t)(2 * cp + 1) * HH * WW]);
    packed[cp] = (unsigned)lo | ((unsigned)hi << 16);
  }
  uint4* __restrict__ dst = (uint4*)(xT + ((size_t)bh * WW + w) * CIN);
  dst[0] = make_uint4(packed[0], packed[1], packed[2], packed[3]);
  dst[1] = make_uint4(packed[4], packed[5], packed[6], packed[7]);
}

// ---------------------------------------------------------------------------
// Main: TDM-staged polyphase GEMM.
// ---------------------------------------------------------------------------
__global__ __launch_bounds__(256) void periodic_conv_wmma_tdm_kernel(
    const u16* __restrict__ xT,      // (B, H, W, CIN) bf16
    const u16* __restrict__ Wt2,     // (4, COUT, KPAD) bf16 tap-major
    float* __restrict__ out)         // (B, COUT, H, W)
{
  // Raw halo tile, channel-minor, unpadded rows (TDM writes contiguously):
  // Raw[hl][wl][c]; row 17 is a zero row for the K-pad tap.
  __shared__ __align__(16) u16 Raw[RAWH + 1][RAWW][CIN];  // 19008 B
  __shared__ __align__(16) u16 Wt[COUT][KPAD];            // 20480 B

  const int bw  = blockIdx.x & 7;
  const int bh  = (blockIdx.x >> 3) & 15;
  const int g   = (blockIdx.x >> 7) & 3;
  const int b   = blockIdx.x >> 9;
  const int gy  = g >> 1;
  const int gx  = g & 1;
  const int hp0 = bh * HT;
  const int wp0 = bw * WT;

  const int tid  = threadIdx.x;
  const int lane = tid & 31;
  const int wave = tid >> 5;

  // ---- Stage weights: straight b128 copies of pre-converted bf16 ----
  {
    const uint4* __restrict__ ws = (const uint4*)(Wt2 + (size_t)g * COUT * KPAD);
    uint4* __restrict__ wd = (uint4*)&Wt[0][0];
#pragma unroll
    for (int i = 0; i < (COUT * KPAD) / 8 / 256; ++i)   // 5 iters
      wd[tid + i * 256] = ws[tid + i * 256];
  }
  // ---- Zero row for padded tap ----
  {
    uint4* __restrict__ zd = (uint4*)&Raw[RAWH][0][0];
    for (int i = tid; i < (RAWW * CIN) / 8; i += 256) {  // 66 vec stores
      zd[i] = make_uint4(0u, 0u, 0u, 0u);
    }
  }

  const int h0 = 2 * hp0 + gy - 1;
  const int w0 = 2 * wp0 + gx - 1;
  const bool interior = (h0 >= 0) && (w0 >= 0);   // block-uniform

  if (interior) {
    // ---- One TDM descriptor loads the whole tile (OOB bottom/right -> 0) ----
    if (wave == 0) {
      const unsigned lds_addr =
          (unsigned)(size_t)(__attribute__((address_space(3))) u16*)&Raw[0][0][0];
      const unsigned long long ga =
          (unsigned long long)(size_t)(xT + (((size_t)b * HH + h0) * WW + w0) * CIN);
      const unsigned td0 = (unsigned)((WW - w0) * CIN);  // valid elems left in row
      const unsigned td1 = (unsigned)(HH - h0);          // valid rows left
      const unsigned tile0 = RAWW * CIN;                 // 528 elements per row
      const unsigned tile1 = RAWH;                       // 17 rows
      const unsigned long long stride0 = (unsigned long long)WW * CIN;  // 4096

      u32x4 g0 = { 1u,                                   // count=1, user D#
                   lds_addr,
                   (unsigned)(ga & 0xFFFFFFFFull),
                   (unsigned)((ga >> 32) & 0x1FFFFFFull) | 0x80000000u };  // type=2
      i32x8 g1 = { (int)(1u << 16),                      // data_size=2B, mask=0
                   (int)((td0 & 0xFFFFu) << 16),         // dim0[15:0]
                   (int)((td0 >> 16) | ((td1 & 0xFFFFu) << 16)),
                   (int)((td1 >> 16) | (tile0 << 16)),
                   (int)(tile1),                         // tile_dim1 | tile_dim2=0
                   (int)(stride0 & 0xFFFFFFFFull),
                   (int)(stride0 >> 32),                 // | dim1_stride lo16 = 0
                   0 };
      i32x4 gz4 = { 0, 0, 0, 0 };
      i32x8 gz8 = { 0, 0, 0, 0, 0, 0, 0, 0 };
      __builtin_amdgcn_tensor_load_to_lds(g0, g1, gz4, gz4, gz8, 0);
      __builtin_amdgcn_s_wait_tensorcnt(0);
    }
  } else {
    // ---- Border blocks: manual vector copy with zero fill ----
    for (int p = tid; p < RAWH * RAWW; p += 256) {
      const int hl = p / RAWW;
      const int wl = p - hl * RAWW;
      const int h  = h0 + hl;
      const int w  = w0 + wl;
      uint4 lo = make_uint4(0u, 0u, 0u, 0u);
      uint4 hi = lo;
      if ((unsigned)h < (unsigned)HH && (unsigned)w < (unsigned)WW) {
        const uint4* __restrict__ s =
            (const uint4*)(xT + (((size_t)b * HH + h) * WW + w) * CIN);
        lo = s[0];
        hi = s[1];
      }
      uint4* __restrict__ d = (uint4*)&Raw[hl][wl][0];
      d[0] = lo;
      d[1] = hi;
    }
  }

  __syncthreads();

  // ---- WMMA phase ----
  const int half = lane >> 4;
  const int row  = lane & 15;
  const int mw   = wave >> 1;
  const int nq   = wave & 1;

  const v8f vzero = {0.f, 0.f, 0.f, 0.f, 0.f, 0.f, 0.f, 0.f};
  v8f acc[4];
#pragma unroll
  for (int j = 0; j < 4; ++j) acc[j] = vzero;

  const u16* __restrict__ arow = &Wt[mw * 16 + row][0];

#pragma unroll
  for (int kc = 0; kc < KPAD / 32; ++kc) {
    const int k0 = kc * 32;
    const v8u a0 = *(const v8u*)(arow + k0 + half * 16);
    const v8u a1 = *(const v8u*)(arow + k0 + half * 16 + 8);
    const v16bf A = frag_cast(a0, a1);

    const int t   = 2 * kc + half;
    const int ky  = t / 3;
    const int kx  = t - 3 * ky;
    const int pad = (t >= 9);

#pragma unroll
    for (int j = 0; j < 4; ++j) {
      const int nh = nq * 4 + j;
      const int hl = pad ? RAWH : (2 * nh + ky);
      const int wl = pad ? row  : (2 * row + kx);
      const u16* __restrict__ bp = &Raw[hl][wl][0];
      const v16bf B = frag_cast(*(const v8u*)bp, *(const v8u*)(bp + 8));
      acc[j] = __builtin_amdgcn_wmma_f32_16x16x32_bf16(
          false, A, false, B, (short)0, acc[j], false, false);
    }
  }

#pragma unroll
  for (int j = 0; j < 4; ++j) {
    const int h = 2 * (hp0 + nq * 4 + j) + gy;
    const int w = 2 * (wp0 + row) + gx;
    float* __restrict__ orow =
        out + (((size_t)b * COUT + mw * 16 + half * 8) * HH + h) * WW + w;
#pragma unroll
    for (int r = 0; r < 8; ++r) {
      __builtin_nontemporal_store(acc[j][r], orow + (size_t)r * HH * WW);
    }
  }
}

// ---------------------------------------------------------------------------
// Fallback (self-contained, no workspace): round-2 kernel.
// ---------------------------------------------------------------------------
__global__ __launch_bounds__(256) void periodic_conv_wmma_fallback_kernel(
    const float* __restrict__ x, const float* __restrict__ Wm,
    float* __restrict__ out)
{
  __shared__ __align__(16) u16 Raw[RAWH + 1][RAWW + 1][CIN];
  __shared__ __align__(16) u16 Wt[COUT][KPAD];

  const int bw  = blockIdx.x & 7;
  const int bh  = (blockIdx.x >> 3) & 15;
  const int g   = (blockIdx.x >> 7) & 3;
  const int b   = blockIdx.x >> 9;
  const int gy  = g >> 1;
  const int gx  = g & 1;
  const int hp0 = bh * HT;
  const int wp0 = bw * WT;

  const int tid = threadIdx.x;
  const float* __restrict__ Wg = Wm + (size_t)g * COUT * KC;

  for (int i = tid; i < COUT * KPAD; i += 256) {
    const int co = i / KPAD;
    const int k  = i - co * KPAD;
    const int t  = k >> 4;
    const int c  = k & 15;
    float v = (t < 9) ? Wg[co * KC + c * 9 + t] : 0.0f;
    Wt[co][k] = f32_to_bf16_rne(v);
  }

  const int h0 = 2 * hp0 + gy - 1;
  const int w0 = 2 * wp0 + gx - 1;
  for (int i = tid; i < CIN * RAWH * RAWW; i += 256) {
    const int c  = i / (RAWH * RAWW);
    const int r  = i - c * (RAWH * RAWW);
    const int hl = r / RAWW;
    const int wl = r - hl * RAWW;
    const int h  = h0 + hl;
    const int w  = w0 + wl;
    float v = 0.0f;
    if ((unsigned)h < (unsigned)HH && (unsigned)w < (unsigned)WW) {
      v = x[(((size_t)b * CIN + c) * HH + h) * WW + w];
    }
    Raw[hl][wl][c] = f32_to_bf16_rne(v);
  }
  for (int i = tid; i < (RAWW + 1) * CIN; i += 256) (&Raw[RAWH][0][0])[i] = 0;

  __syncthreads();

  const int lane = tid & 31;
  const int wave = tid >> 5;
  const int half = lane >> 4;
  const int row  = lane & 15;
  const int mw   = wave >> 1;
  const int nq   = wave & 1;

  const v8f vzero = {0.f, 0.f, 0.f, 0.f, 0.f, 0.f, 0.f, 0.f};
  v8f acc[4];
#pragma unroll
  for (int j = 0; j < 4; ++j) acc[j] = vzero;

  const u16* __restrict__ arow = &Wt[mw * 16 + row][0];

#pragma unroll
  for (int kc = 0; kc < KPAD / 32; ++kc) {
    const int k0 = kc * 32;
    const v16bf A = frag_cast(*(const v8u*)(arow + k0 + half * 16),
                              *(const v8u*)(arow + k0 + half * 16 + 8));
    const int t   = 2 * kc + half;
    const int ky  = t / 3;
    const int kx  = t - 3 * ky;
    const int pad = (t >= 9);
#pragma unroll
    for (int j = 0; j < 4; ++j) {
      const int nh = nq * 4 + j;
      const int hl = pad ? RAWH : (2 * nh + ky);
      const int wl = pad ? row  : (2 * row + kx);
      const u16* __restrict__ bp = &Raw[hl][wl][0];
      const v16bf B = frag_cast(*(const v8u*)bp, *(const v8u*)(bp + 8));
      acc[j] = __builtin_amdgcn_wmma_f32_16x16x32_bf16(
          false, A, false, B, (short)0, acc[j], false, false);
    }
  }

#pragma unroll
  for (int j = 0; j < 4; ++j) {
    const int h = 2 * (hp0 + nq * 4 + j) + gy;
    const int w = 2 * (wp0 + row) + gx;
    float* __restrict__ orow =
        out + (((size_t)b * COUT + mw * 16 + half * 8) * HH + h) * WW + w;
#pragma unroll
    for (int r = 0; r < 8; ++r)
      __builtin_nontemporal_store(acc[j][r], orow + (size_t)r * HH * WW);
  }
}

extern "C" void kernel_launch(void* const* d_in, const int* in_sizes, int n_in,
                              void* d_out, int out_size, void* d_ws, size_t ws_size,
                              hipStream_t stream) {
  (void)in_sizes; (void)n_in; (void)out_size;
  const float* x  = (const float*)d_in[0];
  const float* Wm = (const float*)d_in[1];
  float* out = (float*)d_out;

  const size_t xt_elems = (size_t)BATCH * HH * WW * CIN;
  const size_t w_elems  = (size_t)4 * COUT * KPAD;
  const size_t need = (xt_elems + w_elems) * sizeof(u16);

  if (d_ws != nullptr && ws_size >= need) {
    u16* xT  = (u16*)d_ws;
    u16* Wt2 = xT + xt_elems;
    hipLaunchKernelGGL(periodic_conv_prep_kernel, dim3(BATCH * HH + 1), dim3(256),
                       0, stream, x, Wm, xT, Wt2);
    hipLaunchKernelGGL(periodic_conv_wmma_tdm_kernel, dim3(4096), dim3(256),
                       0, stream, xT, Wt2, out);
  } else {
    hipLaunchKernelGGL(periodic_conv_wmma_fallback_kernel, dim3(4096), dim3(256),
                       0, stream, x, Wm, out);
  }
}
